// Graph_module_net_0_loss_18631568130083
// MI455X (gfx1250) — matrix operand order repaired and checked by reading, not compiled, not run on hardware
//
#include <hip/hip_runtime.h>
#include <hip/hip_bf16.h>

// ---------------------------------------------------------------------------
// MI455X (gfx1250) implementation.
//  - Heavy matmuls (gts GEMM, o1m/o2m attention aggregation) use
//    v_wmma_f32_16x16x32_bf16 (wave32 WMMA, f32 accumulate).
//  - attn tensor stored bf16 (64MiB) in [b][h][i][j] layout: rows are
//    contiguous for top-k and lane-contiguous for WMMA B-fragments.
//  - gconv epilogues emit a bf16 mirror of o1/o2 so the aggregation GEMM
//    loads A-fragments directly (no LDS, no per-iteration cvt VALU).
//  - k_gts stages fp32 A tiles to LDS via global_load_async_to_lds_b128
//    (ASYNCcnt + s_wait_asynccnt), with a sync fallback.
// Workspace layout (bytes):
//   0          attn bf16 (64MiB)
//   +64MiB     o1   fp32 16MiB  [b][mid][n]
//   +80MiB     o1m  fp32 16MiB
//   +96MiB     o2   fp32 16MiB
//   +112MiB    o2m  fp32 16MiB
//   +128MiB    o1bf bf16  8MiB
//   +136MiB    o2bf bf16  8MiB
//   +144MiB    pj(128KiB), pi(128KiB), col bitmask(128B)   (~151.3MiB total)
// ---------------------------------------------------------------------------

typedef __attribute__((ext_vector_type(16))) __bf16          v16bf;
typedef __attribute__((ext_vector_type(8)))  float           v8f;
typedef __attribute__((ext_vector_type(16))) unsigned short  v16us;
typedef __attribute__((ext_vector_type(8)))  unsigned short  v8us;
typedef __attribute__((vector_size(4 * sizeof(int)))) int    v4i_t;

union BFrag { v16us us; v16bf bf; unsigned short h[16]; };
union AFrag { v16bf bf; struct { v8us lo; v8us hi; } s; };

__device__ __forceinline__ unsigned short f2bf(float f) {
  union { float f; unsigned u; } c; c.f = f;
  unsigned u = c.u;
  unsigned r = (u + 0x7FFFu + ((u >> 16) & 1u)) >> 16;   // RNE
  return (unsigned short)r;
}
__device__ __forceinline__ float bf2f(unsigned short h) {
  union { unsigned u; float f; } c; c.u = ((unsigned)h) << 16; return c.f;
}

__device__ __forceinline__ int   wredi(int v)   { for (int i = 16; i > 0; i >>= 1) v += __shfl_xor(v, i, 32); return v; }
__device__ __forceinline__ float wredf(float v) { for (int i = 16; i > 0; i >>= 1) v += __shfl_xor(v, i, 32); return v; }

// ------------------------- async copy helpers ------------------------------
#if defined(__HIP_DEVICE_COMPILE__) && __has_builtin(__builtin_amdgcn_global_load_async_to_lds_b128)
#define ATHENA_ASYNC 1
#endif

__device__ __forceinline__ void async_copy16(float* lds_dst, const float* gsrc) {
#if defined(ATHENA_ASYNC)
  __builtin_amdgcn_global_load_async_to_lds_b128(
      (__attribute__((address_space(1))) v4i_t*)gsrc,
      (__attribute__((address_space(3))) v4i_t*)lds_dst, 0, 0);
#else
  *(float4*)lds_dst = *(const float4*)gsrc;
#endif
}
__device__ __forceinline__ void wait_async() {
#if defined(ATHENA_ASYNC)
#if __has_builtin(__builtin_amdgcn_s_wait_asynccnt)
  __builtin_amdgcn_s_wait_asynccnt(0);
#else
  asm volatile("s_wait_asynccnt 0" ::: "memory");
#endif
#endif
}

// --------------------------- tiny utility kernels --------------------------
__global__ void k_zero32(unsigned* col) { col[threadIdx.x] = 0u; }

// pj[b,n,h] = input[b,n,:].W_attn[h,:C] ; pi uses W_attn[h,C:]
__global__ void k_pjpi(const float* __restrict__ input, const float* __restrict__ W_attn,
                       float* __restrict__ pj, float* __restrict__ pi) {
  int gid  = blockIdx.x * 256 + threadIdx.x;        // 65536
  int which = gid & 1;
  int h     = (gid >> 1) & 3;
  int row   = gid >> 3;                             // b*1024+n
  const float* x = input + (size_t)row * 256;
  const float* w = W_attn + h * 512 + which * 256;
  float s = 0.f;
#pragma unroll 8
  for (int c = 0; c < 256; c++) s += x[c] * w[c];
  (which ? pi : pj)[row * 4 + h] = s;
}

// raw attn[b][h][i][j] = sigmoid(pj[b,j,h]+pi[b,i,h]+b_attn[h]) * roi  (bf16)
__global__ void k_attn_raw(const float* __restrict__ pj, const float* __restrict__ pi,
                           const float* __restrict__ b_attn, const float* __restrict__ roi,
                           const int* __restrict__ sm, unsigned short* __restrict__ attn) {
  size_t gid = (size_t)blockIdx.x * 256 + threadIdx.x;   // 8*4*1024*256
  int    j0  = (int)(gid & 255) * 4;
  size_t r   = gid >> 8;                                 // (b*4+h)*1024 + i
  int i  = (int)(r & 1023);
  int bh = (int)(r >> 10);
  int h = bh & 3, b = bh >> 2;
  float pii = pi[(b * 1024 + i) * 4 + h];
  float ba  = b_attn[h];
  const float* roirow = roi + ((size_t)b * 1024 + i) * 1024 + j0;
  const int*   smrow  = sm + b * 1024 + j0;
  ushort4 ov;
  unsigned short t[4];
#pragma unroll
  for (int q = 0; q < 4; q++) {
    int j = j0 + q;
    float x = pj[(b * 1024 + j) * 4 + h] + pii + ba;
    float s = 1.0f / (1.0f + __expf(-x));
    t[q] = f2bf(s * roirow[q] * (float)smrow[q]);
  }
  ov.x = t[0]; ov.y = t[1]; ov.z = t[2]; ov.w = t[3];
  *(ushort4*)(attn + r * 1024 + j0) = ov;
}

// exact top-64 marking (strict > threshold, then lowest-index ties) for one
// wave32-owned row of 1024 nonneg bf16 keys (uint16 order == float order).
__device__ void topk_mark(const unsigned short v[32], int lane, unsigned* smask, int inv) {
  int lo = 0, hi = 65536;
  while (hi - lo > 1) {                         // 16 iterations
    int mid = (lo + hi) >> 1;
    int c = 0;
#pragma unroll
    for (int t = 0; t < 32; t++) {
      int k = inv ? (0xFFFF - (int)v[t]) : (int)v[t];
      c += (k >= mid);
    }
    c = wredi(c);
    if (c >= 64) lo = mid; else hi = mid;
  }
  int thr = lo;
  int cgt = 0, ctie = 0;
#pragma unroll
  for (int t = 0; t < 32; t++) {
    int k = inv ? (0xFFFF - (int)v[t]) : (int)v[t];
    cgt  += (k > thr);
    ctie += (k == thr);
  }
  int totgt = wredi(cgt);
  for (int t = 0; t < 32; t++) {
    int k = inv ? (0xFFFF - (int)v[t]) : (int)v[t];
    if (k > thr) { int j = lane * 32 + t; atomicOr(&smask[j >> 5], 1u << (j & 31)); }
  }
  int need = 64 - totgt;
  int incl = ctie;
  for (int d = 1; d < 32; d <<= 1) { int n = __shfl_up(incl, d, 32); if (lane >= d) incl += n; }
  int excl = incl - ctie;
  int budget = need - excl; budget = budget < 0 ? 0 : budget; budget = budget > ctie ? ctie : budget;
  int used = 0;
  for (int t = 0; t < 32 && used < budget; t++) {
    int k = inv ? (0xFFFF - (int)v[t]) : (int)v[t];
    if (k == thr) { int j = lane * 32 + t; atomicOr(&smask[j >> 5], 1u << (j & 31)); used++; }
  }
}

__global__ void k_topk(const unsigned short* __restrict__ attn, unsigned* __restrict__ colmask) {
  __shared__ unsigned smask[32];
  int tid = threadIdx.x;
  if (tid < 32) smask[tid] = 0u;
  __syncthreads();
  int lane = tid & 31, w = tid >> 5;
  size_t row = (size_t)blockIdx.x * 8 + w;           // (b*4+h)*1024 + i
  const unsigned short* p = attn + row * 1024 + lane * 32;
  unsigned short v[32];
  v16us a0 = ((const v16us*)p)[0], a1 = ((const v16us*)p)[1];
#pragma unroll
  for (int t = 0; t < 16; t++) { v[t] = a0[t]; v[16 + t] = a1[t]; }
  topk_mark(v, lane, smask, 0);   // top 64
  topk_mark(v, lane, smask, 1);   // bottom 64
  __syncthreads();
  if (tid < 32 && smask[tid]) atomicOr(&colmask[tid], smask[tid]);
}

// attn = (attn*col[j] + (i==j && sm[b,i]==0)) / 128  (in-place, bf16)
__global__ void k_attn_final(unsigned short* __restrict__ attn,
                             const unsigned* __restrict__ colmask,
                             const int* __restrict__ sm) {
  size_t gid = (size_t)blockIdx.x * 256 + threadIdx.x;  // 4.19M threads, 8 elems each
  size_t idx = gid * 8;
  size_t r   = idx >> 10;
  int j0 = (int)(idx & 1023);
  int i  = (int)(r & 1023);
  int b  = (int)(r >> 12);
  int smz = (sm[b * 1024 + i] == 0);
  ushort4 a = ((ushort4*)(attn + idx))[0];
  ushort4 c = ((ushort4*)(attn + idx))[1];
  unsigned short e[8] = {a.x, a.y, a.z, a.w, c.x, c.y, c.z, c.w};
#pragma unroll
  for (int q = 0; q < 8; q++) {
    int j = j0 + q;
    float vv = bf2f(e[q]);
    vv *= (float)((colmask[j >> 5] >> (j & 31)) & 1u);
    if (smz && j == i) vv += 1.0f;
    e[q] = f2bf(vv * (1.0f / 128.0f));
  }
  a.x = e[0]; a.y = e[1]; a.z = e[2]; a.w = e[3];
  c.x = e[4]; c.y = e[5]; c.z = e[6]; c.w = e[7];
  ((ushort4*)(attn + idx))[0] = a;
  ((ushort4*)(attn + idx))[1] = c;
}

// grouped conv + relu: y[b][m][n] = relu(sum_c w[g][o][c]*x(b, g*Cg+c, n) + bias)
// also emits a bf16 mirror (ybf) consumed as WMMA A-matrix by k_agg.
__global__ void k_gconv(const float* __restrict__ x, const float* __restrict__ wgt,
                        const float* __restrict__ bias, float* __restrict__ y,
                        unsigned short* __restrict__ ybf,
                        int Cg, size_t sB, int sC, int sN) {
  int m = blockIdx.x & 511, b = blockIdx.x >> 9;
  int g = m >> 7, o = m & 127;
  const float* wr = wgt + ((size_t)g * 128 + o) * Cg;
  float bb = bias[g * 128 + o];
  int tid = threadIdx.x;
  float acc[4] = {bb, bb, bb, bb};
  const float* xb = x + (size_t)b * sB + (size_t)(g * Cg) * sC;
  for (int c = 0; c < Cg; c++) {
    float wv = wr[c];
    const float* xc = xb + (size_t)c * sC;
#pragma unroll
    for (int q = 0; q < 4; q++) acc[q] += wv * xc[(size_t)(tid + q * 256) * sN];
  }
  size_t rowo = ((size_t)b * 512 + m) * 1024;
#pragma unroll
  for (int q = 0; q < 4; q++) {
    float v = fmaxf(acc[q], 0.f);
    y[rowo + tid + q * 256]   = v;
    ybf[rowo + tid + q * 256] = f2bf(v);
  }
}

// WMMA aggregation: OM[bh][c][i] = sum_j X[bh][c][j] * attn[bh][i][j]
// A = bf16 mirror of o1/o2 (lane-contiguous), B = attn bf16. No LDS needed.
__global__ void k_agg(const unsigned short* __restrict__ Xbf,
                      const unsigned short* __restrict__ attn,
                      float* __restrict__ OM) {
  int tid = threadIdx.x, lane = tid & 31, w = tid >> 5;
  int wm = (w & 1) * 16, wn = (w >> 1) * 16;
  int bh = blockIdx.z;
  int m0 = blockIdx.x * 32, n0 = blockIdx.y * 64;
  int la = lane & 15, lh = lane >> 4;
  // A row owned by this lane (contiguous K order matching B: low lanes K 0..15;
  // high lanes K 16..31 -> two 16B chunks at +lh*16 and +lh*16+8)
  const unsigned short* Arow = Xbf + ((size_t)bh * 128 + m0 + wm + la) * 1024;
  const unsigned short* Brow = attn + (size_t)bh * 1024 * 1024 + (size_t)(n0 + wn + la) * 1024;
  v8f acc = {0, 0, 0, 0, 0, 0, 0, 0};
  for (int k0 = 0; k0 < 1024; k0 += 32) {
    AFrag af;
    af.s.lo = *(const v8us*)(Arow + k0 + lh * 16);
    af.s.hi = *(const v8us*)(Arow + k0 + lh * 16 + 8);
    BFrag bfr;
    const unsigned short* bp = Brow + k0 + lh * 16;
    bfr.us = *(const v16us*)bp;
    if (k0 + 32 < 1024) {
      __builtin_prefetch(bp + 32, 0, 1);          // global_prefetch next B chunk
      __builtin_prefetch(Arow + k0 + 32, 0, 1);   // and next A chunk
    }
    acc = __builtin_amdgcn_wmma_f32_16x16x32_bf16(false, af.bf, false, bfr.bf,
                                                  (short)0, acc, false, false);
  }
  float* out = OM + ((size_t)bh * 128 + m0 + wm) * 1024 + (n0 + wn + la);
#pragma unroll
  for (int rr = 0; rr < 8; rr++) out[(size_t)(rr + lh * 8) * 1024] = acc[rr];
}

// WMMA GEMM + bias + relu: gts[row][out] = relu(gt_feat[row,:].W_gt[out,:] + b_gt)
// fp32 A staged to LDS with async-to-LDS DMA, converted to bf16 fragments.
__global__ void k_gts(const float* __restrict__ A, const float* __restrict__ W,
                      const float* __restrict__ bias, float* __restrict__ Y) {
  __shared__ float As[32 * 36];
  int tid = threadIdx.x, lane = tid & 31, w = tid >> 5;
  int wm = (w & 1) * 16, wn = (w >> 1) * 16;
  int m0 = blockIdx.x * 32, n0 = blockIdx.y * 64;
  int r = tid >> 3, s = tid & 7;
  int la = lane & 15, lh = lane >> 4;
  v8f acc = {0, 0, 0, 0, 0, 0, 0, 0};
  for (int k0 = 0; k0 < 256; k0 += 32) {
    __syncthreads();
    async_copy16(&As[r * 36 + s * 4], A + (size_t)(m0 + r) * 256 + k0 + s * 4);
    wait_async();
    __syncthreads();
    BFrag af;
    const float* ar = &As[(wm + la) * 36 + lh * 16];
#pragma unroll
    for (int t = 0; t < 8; t++) { af.h[t] = f2bf(ar[t]); af.h[8 + t] = f2bf(ar[8 + t]); }
    BFrag bfr;
    const float* wr = W + (size_t)(n0 + wn + la) * 256 + k0 + lh * 16;
#pragma unroll
    for (int t = 0; t < 16; t++) bfr.h[t] = f2bf(wr[t]);
    acc = __builtin_amdgcn_wmma_f32_16x16x32_bf16(false, af.bf, false, bfr.bf,
                                                  (short)0, acc, false, false);
  }
  int col = n0 + wn + la;
  float bb = bias[col];
#pragma unroll
  for (int rr = 0; rr < 8; rr++) {
    int row = m0 + wm + rr + lh * 8;
    Y[(size_t)row * 512 + col] = fmaxf(acc[rr] + bb, 0.f);
  }
}

// o1 += LN(o1m) across MID channels (per (b,i)); OM layout [b][mid][i]
__global__ void k_ln_add(const float* __restrict__ OM, const float* __restrict__ gam,
                         const float* __restrict__ bet, float* __restrict__ O) {
  int bi = blockIdx.x, b = bi >> 10, i = bi & 1023;
  const float* xm = OM + (size_t)b * 524288 + i;
  int tid = threadIdx.x, lane = tid & 31, w = tid >> 5;
  float x[4], s1 = 0.f, s2 = 0.f;
#pragma unroll
  for (int q = 0; q < 4; q++) { float v = xm[(size_t)(tid + q * 128) * 1024]; x[q] = v; s1 += v; s2 += v * v; }
  s1 = wredf(s1); s2 = wredf(s2);
  __shared__ float red[8];
  if (lane == 0) { red[w] = s1; red[4 + w] = s2; }
  __syncthreads();
  float S1 = red[0] + red[1] + red[2] + red[3];
  float S2 = red[4] + red[5] + red[6] + red[7];
  float mu = S1 * (1.f / 512.f);
  float rin = rsqrtf(S2 * (1.f / 512.f) - mu * mu + 1e-6f);
  float* orow = O + (size_t)b * 524288 + i;
#pragma unroll
  for (int q = 0; q < 4; q++) {
    int m = tid + q * 128;
    orow[(size_t)m * 1024] += (x[q] - mu) * rin * gam[m] + bet[m];
  }
}

// y = LN(o2m); node_feat[b,i,:] = y ; output2[b,i,:] = o2[b,:,i] + y
__global__ void k_ln2(const float* __restrict__ OM, const float* __restrict__ O2,
                      const float* __restrict__ gam, const float* __restrict__ bet,
                      float* __restrict__ out2, float* __restrict__ nodef) {
  int bi = blockIdx.x, b = bi >> 10, i = bi & 1023;
  const float* xm = OM + (size_t)b * 524288 + i;
  int tid = threadIdx.x, lane = tid & 31, w = tid >> 5;
  float x[4], s1 = 0.f, s2 = 0.f;
#pragma unroll
  for (int q = 0; q < 4; q++) { float v = xm[(size_t)(tid + q * 128) * 1024]; x[q] = v; s1 += v; s2 += v * v; }
  s1 = wredf(s1); s2 = wredf(s2);
  __shared__ float red[8];
  if (lane == 0) { red[w] = s1; red[4 + w] = s2; }
  __syncthreads();
  float S1 = red[0] + red[1] + red[2] + red[3];
  float S2 = red[4] + red[5] + red[6] + red[7];
  float mu = S1 * (1.f / 512.f);
  float rin = rsqrtf(S2 * (1.f / 512.f) - mu * mu + 1e-6f);
  size_t rowo = ((size_t)b * 1024 + i) * 512;
#pragma unroll
  for (int q = 0; q < 4; q++) {
    int m = tid + q * 128;
    float y = (x[q] - mu) * rin * gam[m] + bet[m];
    nodef[rowo + m] = y;
    out2[rowo + m] = y + O2[(size_t)b * 524288 + (size_t)m * 1024 + i];
  }
}

// ---------------------------------------------------------------------------
extern "C" void kernel_launch(void* const* d_in, const int* in_sizes, int n_in,
                              void* d_out, int out_size, void* d_ws, size_t ws_size,
                              hipStream_t stream) {
  const float* input  = (const float*)d_in[0];
  const float* roi    = (const float*)d_in[1];
  const int*   sm     = (const int*)  d_in[2];
  const float* gtf    = (const float*)d_in[3];
  const float* W_attn = (const float*)d_in[4];
  const float* b_attn = (const float*)d_in[5];
  const float* W1g    = (const float*)d_in[6];
  const float* b1g    = (const float*)d_in[7];
  const float* W2g    = (const float*)d_in[8];
  const float* b2g    = (const float*)d_in[9];
  const float* ln1_g  = (const float*)d_in[10];
  const float* ln1_b  = (const float*)d_in[11];
  const float* ln2_g  = (const float*)d_in[12];
  const float* ln2_b  = (const float*)d_in[13];
  const float* W_gt   = (const float*)d_in[14];
  const float* b_gt   = (const float*)d_in[15];

  char* ws = (char*)d_ws;
  unsigned short* attnB = (unsigned short*)(ws);                   //  64 MiB bf16
  float*          o1    = (float*)(ws + (size_t)67108864);         //  16 MiB
  float*          o1m   = (float*)(ws + (size_t)83886080);         //  16 MiB
  float*          o2    = (float*)(ws + (size_t)100663296);        //  16 MiB
  float*          o2m   = (float*)(ws + (size_t)117440512);        //  16 MiB
  unsigned short* o1bf  = (unsigned short*)(ws + (size_t)134217728); // 8 MiB
  unsigned short* o2bf  = (unsigned short*)(ws + (size_t)142606336); // 8 MiB
  float*          pj    = (float*)(ws + (size_t)150994944);
  float*          pi    = (float*)(ws + (size_t)151126016);
  unsigned*       col   = (unsigned*)(ws + (size_t)151257088);

  float* out2  = (float*)d_out;                 // (B,N,OUT)
  float* gts   = (float*)d_out + 4194304;       // (B,N,OUT)
  float* nodef = (float*)d_out + 8388608;       // (B,N,OUT)

  k_zero32<<<1, 32, 0, stream>>>(col);
  k_pjpi<<<256, 256, 0, stream>>>(input, W_attn, pj, pi);
  k_gts<<<dim3(256, 8), 256, 0, stream>>>(gtf, W_gt, b_gt, gts);
  k_attn_raw<<<32768, 256, 0, stream>>>(pj, pi, b_attn, roi, sm, attnB);
  k_topk<<<4096, 256, 0, stream>>>(attnB, col);
  k_attn_final<<<16384, 256, 0, stream>>>(attnB, col, sm);
  // gconv1: x = input[b][n][c] -> strides (sB=1024*256, sC=1, sN=256)
  k_gconv<<<4096, 256, 0, stream>>>(input, W1g, b1g, o1, o1bf, 64, (size_t)262144, 1, 256);
  k_agg<<<dim3(4, 16, 32), 256, 0, stream>>>(o1bf, attnB, o1m);
  k_ln_add<<<8192, 128, 0, stream>>>(o1m, ln1_g, ln1_b, o1);
  // gconv2: x = o1[b][mid][n] -> strides (sB=512*1024, sC=1024, sN=1)
  k_gconv<<<4096, 256, 0, stream>>>(o1, W2g, b2g, o2, o2bf, 128, (size_t)524288, 1024, 1);
  k_agg<<<dim3(4, 16, 32), 256, 0, stream>>>(o2bf, attnB, o2m);
  k_ln2<<<8192, 128, 0, stream>>>(o2m, o2, ln2_g, ln2_b, out2, nodef);
}
